// MwpBertModel_CLS_classfier_90400471646774
// MI455X (gfx1250) — matrix operand order repaired and checked by reading, not compiled, not run on hardware
//
#include <hip/hip_runtime.h>
#include <hip/hip_bf16.h>

// ---------------------------------------------------------------------------
// MwpBertModel CLS head on MI455X (gfx1250, wave32).
// GEMMs: bf16 WMMA (v_wmma_f32_16x16x32_bf16), f32 accumulate.
// 8 wave32s / WG; wave tile 32Mx32N (4 accumulators); double-buffered LDS.
// ---------------------------------------------------------------------------

typedef __attribute__((ext_vector_type(16))) __bf16 v16bf;
typedef __attribute__((ext_vector_type(8)))  float  v8f;

#define Bsz 64
#define Ssz 512
#define Psz 32
#define Lsz 30
#define Hsz 768
#define MROWS (Bsz * Psz)        // 2048

// LDS row stride (elements). 40*2B = 80B: multiple of 16B (b128-aligned
// fragment loads) and staggers banks across rows.
#define LDSTR 40

// ---------------- f32 -> bf16 convert ----------------
__global__ void k_f32_to_bf16(const float* __restrict__ in,
                              __bf16* __restrict__ out, int n) {
    int i = blockIdx.x * blockDim.x + threadIdx.x;
    if (i < n) out[i] = (__bf16)in[i];
}

// ---------------- W3 [512,30] -> padded bf16 [512,32] ----------------
__global__ void k_pad_w3(const float* __restrict__ w3,
                         __bf16* __restrict__ out) {
    int i = blockIdx.x * blockDim.x + threadIdx.x;   // over 512*32
    if (i >= 512 * 32) return;
    int k = i >> 5, n = i & 31;
    out[i] = (n < Lsz) ? (__bf16)w3[k * Lsz + n] : (__bf16)0.0f;
}

// ---------------- gather + concat -> feat bf16 [2048,1536] ----------------
__global__ void k_gather_feat(const float* __restrict__ tok,   // [B,S,H]
                              const float* __restrict__ pool,  // [B,H]
                              const int*   __restrict__ posi,  // [B,P]
                              __bf16* __restrict__ feat) {     // [2048, 2H]
    int r = blockIdx.x;                   // 0..2047
    int b = r >> 5, p = r & 31;
    int pos = posi[b * Psz + p];
    const float* src = tok + ((size_t)b * Ssz + pos) * Hsz;
    const float* pl  = pool + (size_t)b * Hsz;
    __bf16* dst = feat + (size_t)r * (2 * Hsz);
    for (int c = threadIdx.x; c < Hsz; c += blockDim.x) {
        dst[c]       = (__bf16)src[c];
        dst[Hsz + c] = (__bf16)pl[c];
    }
}

// ---------------- generic WMMA GEMM + bias ----------------
// A: [2048, K] bf16 row-major.  Bw: [K, N] bf16 row-major.  bias: [Nbias] f32.
// WN = wave columns: WN=2 -> WG tile 128Mx64N; WN=1 -> WG tile 256Mx32N.
// Wave tile: 32M x 32N (acc[2][2] of 16x16). K stepped by 32, double-buffered.
template <int WN>
__global__ void __launch_bounds__(256)
k_wmma_gemm_bias(const __bf16* __restrict__ A,
                 const __bf16* __restrict__ Bw,
                 const float*  __restrict__ bias,
                 __bf16* __restrict__ out_bf,   // one of out_bf/out_f is null
                 float*  __restrict__ out_f,
                 int K, int N, int Nbias) {
    constexpr int MT    = (8 / WN) * 32;   // WG M tile: 128 or 256
    constexpr int NT    = WN * 32;         // WG N tile: 64 or 32
    constexpr int NCH_A = MT / 64;         // b128 A-chunks per thread (2 or 4)
    constexpr int NCH_B = NT / 32;         // uint2 B-chunks per thread (2 or 1)

    __shared__ __bf16 sA [2][MT * LDSTR];  // A tile [MT rows][32 k]
    __shared__ __bf16 sBt[2][NT * LDSTR];  // B tile transposed [NT n][32 k]

    const int tid  = threadIdx.x;
    const int lane = tid & 31;
    const int wave = tid >> 5;
    const int half = lane >> 4;            // 0: lanes 0-15, 1: lanes 16-31
    const int l16  = lane & 15;
    const int wm   = wave / WN;            // wave row    (0..8/WN-1)
    const int wn   = wave % WN;            // wave column (0..WN-1)

    const int m0 = blockIdx.x * MT;
    const int n0 = blockIdx.y * NT;

    uint4 ra[NCH_A];
    uint2 rb[NCH_B];

    // global -> regs for K-step at k0 (coalesced b128 / b64)
    auto gload = [&](int k0) {
#pragma unroll
        for (int j = 0; j < NCH_A; ++j) {
            const int c = tid + 256 * j;           // row = c/4, grp = c%4
            ra[j] = *(const uint4*)(A + (size_t)(m0 + (c >> 2)) * K +
                                    k0 + (c & 3) * 8);
        }
#pragma unroll
        for (int j = 0; j < NCH_B; ++j) {
            const int c  = tid + 256 * j;          // covers 32k x NT n / 4
            const int k  = c / (NT / 4);
            const int ng = (c % (NT / 4)) * 4;
            rb[j] = *(const uint2*)(Bw + (size_t)(k0 + k) * N + n0 + ng);
        }
    };

    // regs -> LDS buffer `bi` (B transposed to [n][k])
    auto lstore = [&](int bi) {
#pragma unroll
        for (int j = 0; j < NCH_A; ++j) {
            const int c = tid + 256 * j;
            *(uint4*)(&sA[bi][(c >> 2) * LDSTR + (c & 3) * 8]) = ra[j];
        }
#pragma unroll
        for (int j = 0; j < NCH_B; ++j) {
            const int c  = tid + 256 * j;
            const int k  = c / (NT / 4);
            const int ng = (c % (NT / 4)) * 4;
            union { uint2 u; __bf16 h[4]; } v; v.u = rb[j];
#pragma unroll
            for (int q = 0; q < 4; ++q)
                sBt[bi][(ng + q) * LDSTR + k] = v.h[q];
        }
    };

    v8f acc[2][2] = {};

    gload(0);
    lstore(0);
    __syncthreads();

    int buf = 0;
    for (int k0 = 0; k0 < K; k0 += 32) {
        const bool more = (k0 + 32) < K;
        if (more) gload(k0 + 32);      // prefetch next tile while computing

        // ---- fragments: contiguous k order (k = 16*half + s), same for A and B ----
        // A 16x32 bf16: lanes 0-15 hold K 0..15; lanes 16-31 hold
        // K 16..31 (row M = lane&15).
        v16bf fa0, fa1;
        {
            const __bf16* pa = &sA[buf][(wm * 32 + l16) * LDSTR + half * 16];
            ((uint4*)&fa0)[0] = *(const uint4*)(pa);
            ((uint4*)&fa0)[1] = *(const uint4*)(pa + 8);
            const __bf16* pb = pa + 16 * LDSTR;
            ((uint4*)&fa1)[0] = *(const uint4*)(pb);
            ((uint4*)&fa1)[1] = *(const uint4*)(pb + 8);
        }
        // B 32x16 bf16: lanes 0-15 hold K 0..15, lanes 16-31 hold K 16..31,
        // column N = lane&15.  sBt is [n][k] so 16 K-values are contiguous.
        v16bf fb0, fb1;
        {
            const __bf16* pb = &sBt[buf][(wn * 32 + l16) * LDSTR + half * 16];
            ((uint4*)&fb0)[0] = *(const uint4*)(pb);
            ((uint4*)&fb0)[1] = *(const uint4*)(pb + 8);
            const __bf16* pc = pb + 16 * LDSTR;
            ((uint4*)&fb1)[0] = *(const uint4*)(pc);
            ((uint4*)&fb1)[1] = *(const uint4*)(pc + 8);
        }

        acc[0][0] = __builtin_amdgcn_wmma_f32_16x16x32_bf16(
                        false, fa0, false, fb0, (short)0, acc[0][0], false, false);
        acc[0][1] = __builtin_amdgcn_wmma_f32_16x16x32_bf16(
                        false, fa0, false, fb1, (short)0, acc[0][1], false, false);
        acc[1][0] = __builtin_amdgcn_wmma_f32_16x16x32_bf16(
                        false, fa1, false, fb0, (short)0, acc[1][0], false, false);
        acc[1][1] = __builtin_amdgcn_wmma_f32_16x16x32_bf16(
                        false, fa1, false, fb1, (short)0, acc[1][1], false, false);

        if (more) lstore(buf ^ 1);     // fill the other buffer
        __syncthreads();
        buf ^= 1;
    }

    // ---- epilogue: bias + store ----
    // C/D layout: VGPR i -> M = i + 8*half, N = lane&15.
    const int mwbase = m0 + wm * 32;
    const int nwbase = n0 + wn * 32;
#pragma unroll
    for (int ai = 0; ai < 2; ++ai) {
#pragma unroll
        for (int bj = 0; bj < 2; ++bj) {
            const int col = nwbase + bj * 16 + l16;
            const float bv = (col < Nbias) ? bias[col] : 0.0f;
            const int mb = mwbase + ai * 16 + half * 8;
#pragma unroll
            for (int i = 0; i < 8; ++i) {
                const size_t idx = (size_t)(mb + i) * N + col;
                const float v = acc[ai][bj][i] + bv;
                if (out_bf) out_bf[idx] = (__bf16)v;
                else        out_f[idx]  = v;
            }
        }
    }
}

// ---------------- per-slot losses -> atomic accumulators ----------------
// acc: [0]=bce_sum [1]=mf_sum [2]=count_sq_sum [3]=n_eq [4]=pos_sq_sum
__global__ void k_zero_acc(float* acc) {
    if (threadIdx.x < 8) acc[threadIdx.x] = 0.0f;
}

__global__ void k_loss(const float* __restrict__ logits,   // [2048, 32]
                       const float* __restrict__ labels,   // [B, P*L]
                       float* __restrict__ acc) {
    int r = blockIdx.x * blockDim.x + threadIdx.x;
    if (r >= MROWS) return;
    int b = r >> 5, p = r & 31;
    const float* lg = logits + (size_t)r * 32;
    const float* lb = labels + (size_t)b * (Psz * Lsz) + p * Lsz;

    bool  pad = false;
    float bce = 0.0f, cl = 0.0f;
    int   co = 0, po = -1, pl_ = -1;
#pragma unroll
    for (int l = 0; l < Lsz; ++l) {
        const float x = lg[l];
        const float y = lb[l];
        if (y != -1.0f) pad = true;
        // stable softplus(x) - x*y
        const float sp = fmaxf(x, 0.0f) + log1pf(expf(-fabsf(x)));
        bce += sp - x * y;
        if (x > 0.0f) { if (po < 0) po = l; ++co; }   // sigmoid(x) > 0.5
        cl += y;
        if (y == 1.0f && pl_ < 0) pl_ = l;
    }
    if (pad) {
        atomicAdd(&acc[0], bce);
        atomicAdd(&acc[1], 1.0f);
        const float d = (float)co - cl;
        atomicAdd(&acc[2], d * d);
        if (co == 1 && cl == 1.0f) {
            atomicAdd(&acc[3], 1.0f);
            const float dp = (float)po - (float)pl_;
            atomicAdd(&acc[4], dp * dp);
        }
    }
}

__global__ void k_finalize(const float* __restrict__ acc,
                           float* __restrict__ out) {
    const float bce  = acc[0] / (acc[1] * (float)Lsz);
    const float cntl = acc[2] / (float)MROWS;
    const float posl = (acc[3] > 0.0f) ? acc[4] / fmaxf(acc[3], 1.0f) : 0.0f;
    out[0] = bce + 10.0f * cntl + 5.0f * posl;
}

// ---------------------------------------------------------------------------
static inline size_t align256(size_t x) { return (x + 255) & ~(size_t)255; }

extern "C" void kernel_launch(void* const* d_in, const int* in_sizes, int n_in,
                              void* d_out, int out_size, void* d_ws, size_t ws_size,
                              hipStream_t stream) {
    const float* tok   = (const float*)d_in[0];   // [64,512,768]
    const float* pool  = (const float*)d_in[1];   // [64,768]
    const int*   posi  = (const int*)  d_in[2];   // [64,32]
    const float* labs  = (const float*)d_in[3];   // [64,960]
    const float* W1    = (const float*)d_in[4];   // [1536,1024]
    const float* b1    = (const float*)d_in[5];
    const float* W2    = (const float*)d_in[6];   // [1024,512]
    const float* b2    = (const float*)d_in[7];
    const float* W3    = (const float*)d_in[8];   // [512,30]
    const float* b3    = (const float*)d_in[9];
    float* out = (float*)d_out;

    // workspace layout
    char* ws = (char*)d_ws;
    size_t off = 0;
    __bf16* feat = (__bf16*)(ws + off); off = align256(off + (size_t)MROWS * 1536 * 2);
    __bf16* W1b  = (__bf16*)(ws + off); off = align256(off + (size_t)1536 * 1024 * 2);
    __bf16* W2b  = (__bf16*)(ws + off); off = align256(off + (size_t)1024 * 512 * 2);
    __bf16* W3b  = (__bf16*)(ws + off); off = align256(off + (size_t)512 * 32 * 2);
    __bf16* h1   = (__bf16*)(ws + off); off = align256(off + (size_t)MROWS * 1024 * 2);
    __bf16* h2   = (__bf16*)(ws + off); off = align256(off + (size_t)MROWS * 512 * 2);
    float*  lgts = (float*) (ws + off); off = align256(off + (size_t)MROWS * 32 * 4);
    float*  acc  = (float*) (ws + off); off = align256(off + 8 * 4);
    (void)ws_size; (void)in_sizes; (void)n_in; (void)out_size;

    // 1) precision conversion / padding / gather
    {
        int n1 = 1536 * 1024;
        k_f32_to_bf16<<<(n1 + 255) / 256, 256, 0, stream>>>(W1, W1b, n1);
        int n2 = 1024 * 512;
        k_f32_to_bf16<<<(n2 + 255) / 256, 256, 0, stream>>>(W2, W2b, n2);
        k_pad_w3<<<(512 * 32 + 255) / 256, 256, 0, stream>>>(W3, W3b);
        k_gather_feat<<<MROWS, 256, 0, stream>>>(tok, pool, posi, feat);
        k_zero_acc<<<1, 32, 0, stream>>>(acc);
    }

    // 2) MLP: three WMMA GEMMs (identity activation is faithful to reference)
    k_wmma_gemm_bias<2><<<dim3(MROWS / 128, 1024 / 64), 256, 0, stream>>>(
        feat, W1b, b1, h1, nullptr, 1536, 1024, 1024);
    k_wmma_gemm_bias<2><<<dim3(MROWS / 128, 512 / 64), 256, 0, stream>>>(
        h1, W2b, b2, h2, nullptr, 1024, 512, 512);
    k_wmma_gemm_bias<1><<<dim3(MROWS / 256, 1), 256, 0, stream>>>(
        h2, W3b, b3, nullptr, lgts, 512, 32, Lsz);

    // 3) losses
    k_loss<<<(MROWS + 255) / 256, 256, 0, stream>>>(lgts, labs, acc);
    k_finalize<<<1, 1, 0, stream>>>(acc, out);
}
